// WelchPSDEncoder_42683384988286
// MI455X (gfx1250) — compile-verified, matrix operand order checked by reading, not run on hardware
//
#include <hip/hip_runtime.h>
#include <hip/hip_bf16.h>

typedef __attribute__((ext_vector_type(16))) _Float16 v16h;
typedef __attribute__((ext_vector_type(8)))  float    v8f;
typedef unsigned int v4u __attribute__((ext_vector_type(4)));
typedef int          v8i __attribute__((ext_vector_type(8)));
typedef int          v4i __attribute__((ext_vector_type(4)));

#define WPSD_T      7680
#define WPSD_STEP   128
#define WPSD_NSEG   59
#define WPSD_NK     45      // band bins k = 1..45
#define WPSD_MT     6       // 6 row tiles of 16 -> 96 rows (90 used)
#define WPSD_KS     8       // 256 / 32

// Dynamic LDS partition (base = 0: only dynamic shared is used)
#define COEF_OFF    0
#define COEF_BYTES  (WPSD_MT * WPSD_KS * 32 * 16 * 2)           // 49152
#define XRAW_OFF    COEF_BYTES                                   // 49152
#define XRAW_BYTES  (WPSD_T * 4 + (WPSD_T / 128) * 16)           // 31680 (16B pad / 128 floats)
#define PSD_OFF     (XRAW_OFF + XRAW_BYTES)                      // 80832
#define SMEM_BYTES  (PSD_OFF + 48 * 4)                           // 81024

#if defined(__has_builtin)
#  if __has_builtin(__builtin_amdgcn_tensor_load_to_lds) && \
      __has_builtin(__builtin_amdgcn_s_wait_tensorcnt)
#    define WPSD_TDM 1
#  endif
#endif

union PackU { _Float16 h[2]; unsigned u; };
union FragB { v16h v; unsigned u[8]; };

extern "C" __global__ __launch_bounds__(128, 1)
void welch_psd_wmma(const float* __restrict__ x, float* __restrict__ out)
{
    extern __shared__ __align__(16) char smem[];
    _Float16* coefLds = (_Float16*)(smem + COEF_OFF);
    float*    xRaw    = (float*)(smem + XRAW_OFF);   // padded: idx = n + ((n>>7)<<2)
    float*    psdLds  = (float*)(smem + PSD_OFF);

    const int tid  = threadIdx.x;
    const int lane = tid & 31;
    const int wave = tid >> 5;
    const int ch   = blockIdx.x;                     // ch = b*64 + c
    const float* xc = x + (size_t)ch * WPSD_T;

#if WPSD_TDM
    // ---------- issue TDM load first: DMA overlaps the coef build ----------
    // D# group0: count=1 | lds_addr | global_addr[56:0] | type=2
    // D# group1: data_size=4B, pad_enable, pad_interval=6 (128 DW), pad_amount=3 (4 DW),
    //            tensor_dim0 = tile_dim0 = 7680, tensor_dim1 = tile_dim1 = 1, stride0 = 7680
    if (wave == 0) {
        const unsigned long long ga = (unsigned long long)xc;
        v4u g0;
        g0[0] = 1u;                                             // count=1
        g0[1] = (unsigned)XRAW_OFF;                             // lds_addr
        g0[2] = (unsigned)(ga & 0xFFFFFFFFull);                 // global_addr[31:0]
        g0[3] = (unsigned)((ga >> 32) & 0x1FFFFFFull) | (2u << 30); // ga[56:32] | type=2
        v8i g1;
        g1[0] = (int)((2u << 16) | (1u << 20) | (6u << 22) | (3u << 25));
        g1[1] = (int)((unsigned)WPSD_T << 16);                  // tensor_dim0[15:0] @ [63:48]
        g1[2] = (int)(1u << 16);                                // dim0 hi=0, tensor_dim1=1
        g1[3] = (int)((unsigned)WPSD_T << 16);                  // dim1 hi=0, tile_dim0=7680
        g1[4] = 1;                                              // tile_dim1=1, tile_dim2=0
        g1[5] = WPSD_T;                                         // tensor_dim0_stride lo
        g1[6] = 0;
        g1[7] = 0;
        const v4i gz = {0, 0, 0, 0};
#if __clang_major__ >= 23
        const v8i gz8 = {0, 0, 0, 0, 0, 0, 0, 0};
        __builtin_amdgcn_tensor_load_to_lds(g0, g1, gz, gz, gz8, 0);
#else
        __builtin_amdgcn_tensor_load_to_lds(g0, g1, gz, gz, 0);
#endif
    }
#endif

    // ---------- coefficient fragments (overlaps TDM DMA) ----------
    // Row r (0..95): k = r/2 + 1; even r = Re, odd r = Im; k>45 -> 0.
    // Re: w[n]*cos(2*pi*k*n/256) + (k==1 ? 0.25 : 0)   (detrend fold, W[1]=-64)
    // Im: -w[n]*sin(2*pi*k*n/256)
    for (int slot = tid; slot < WPSD_MT * WPSD_KS * 32; slot += 128) {
        const int l    = slot & 31;
        const int frag = slot >> 5;
        const int m    = frag >> 3;
        const int ksI  = frag & 7;
        const int hh   = l >> 4;
        const int r    = m * 16 + (l & 15);
        const int kf   = (r >> 1) + 1;
        v16h vals;
        #pragma unroll
        for (int j = 0; j < 16; ++j) {
            const int v  = j >> 1;
            const int kk = ((v >= 4) ? 16 : 0) + hh * 8 + (v & 3) * 2 + (j & 1);
            const int K  = ksI * 32 + kk;
            float c = 0.0f;
            if (kf <= WPSD_NK) {
                const float w  = 0.5f - 0.5f * __cosf((float)K * 0.02454369260617026f);
                const int   tI = (kf * K) & 255;
                const float a  = (float)tI * 0.02454369260617026f;   // 2*pi/256
                if ((r & 1) == 0)
                    c = w * __cosf(a) + ((kf == 1) ? 0.25f : 0.0f);
                else
                    c = -w * __sinf(a);
            }
            vals[j] = (_Float16)c;
        }
        *(v16h*)&coefLds[slot * 16] = vals;    // 32B aligned -> ds_store_b128 x2
    }

    if (tid < 48) psdLds[tid] = 0.0f;

#if WPSD_TDM
    if (wave == 0) __builtin_amdgcn_s_wait_tensorcnt(0);
#else
    // ---------- fallback: synchronous padded staging ----------
    for (int q = tid; q < WPSD_T / 4; q += 128) {
        const int n = q * 4;
        const float4 f = ((const float4*)xc)[q];
        const int idx = n + ((n >> 7) << 2);
        *(float4*)&xRaw[idx] = f;              // 16B aligned (pad is 16B)
    }
#endif
    __syncthreads();

    // ---------- DFT-as-GEMM via WMMA ----------
    const int  h   = lane >> 4;
    const int  seg = wave * 16 + (lane & 15);          // this lane's segment (col N)
    const bool ok  = seg < WPSD_NSEG;

    v8f acc[WPSD_MT];
    const v8f z = {0.f, 0.f, 0.f, 0.f, 0.f, 0.f, 0.f, 0.f};
    #pragma unroll
    for (int m = 0; m < WPSD_MT; ++m) acc[m] = z;

    #pragma unroll
    for (int ks = 0; ks < WPSD_KS; ++ks) {
        // B fragment (K=32 x N=16): 8x ds_load_b64 + f32->f16 pack, reused by all
        // 6 M tiles. Invalid segments masked branchlessly (EXEC stays full).
        FragB B;
        #pragma unroll
        for (int v = 0; v < 8; ++v) {
            const int kk  = ((v >= 4) ? 16 : 0) + h * 8 + (v & 3) * 2;
            const int n   = seg * WPSD_STEP + ks * 32 + kk;
            const int idx = n + ((n >> 7) << 2);       // padded f32 index
            const float2 f2 = *(const float2*)&xRaw[idx];
            PackU p;
            p.h[0] = (_Float16)f2.x;
            p.h[1] = (_Float16)f2.y;
            B.u[v] = ok ? p.u : 0u;
        }
        #pragma unroll
        for (int m = 0; m < WPSD_MT; ++m) {
            const v16h a = *(const v16h*)&coefLds[((m * WPSD_KS + ks) * 32 + lane) * 16];
            acc[m] = __builtin_amdgcn_wmma_f32_16x16x32_f16(
                false, a, false, B.v, (short)0, acc[m], false, false);
        }
    }

    // ---------- power + reduce over segments ----------
    // C/D layout: VGPR d, lane l -> row M = d + 8*(l>>4). Rows interleaved Re/Im:
    // Re(k)=acc[2t], Im(k)=acc[2t+1] in the SAME lane; k-1 = m*8 + h*4 + t.
    #pragma unroll
    for (int m = 0; m < WPSD_MT; ++m) {
        #pragma unroll
        for (int t = 0; t < 4; ++t) {
            const float re = acc[m][2 * t];
            const float im = acc[m][2 * t + 1];
            float p = re * re + im * im;
            #pragma unroll
            for (int off = 1; off < 16; off <<= 1)     // sum 16 segments
                p += __shfl_xor(p, off, 32);
            if ((lane & 15) == 0) {
                const int kf = m * 8 + h * 4 + t;      // = k-1, 0..47
                if (kf < WPSD_NK) atomicAdd(&psdLds[kf], p);
            }
        }
    }
    __syncthreads();

    // ---------- scale + log1p + store ----------
    if (tid < WPSD_NK) {
        // mult(2) * scale(1/(fs*sum(w^2)) = 1/(256*96)) / nseg(59)
        const float factor = 2.0f / (256.0f * 96.0f * 59.0f);
        out[(size_t)ch * WPSD_NK + tid] = log1pf(psdLds[tid] * factor);
    }
}

extern "C" void kernel_launch(void* const* d_in, const int* in_sizes, int n_in,
                              void* d_out, int out_size, void* d_ws, size_t ws_size,
                              hipStream_t stream) {
    const float* x   = (const float*)d_in[0];
    float*       out = (float*)d_out;
    // One workgroup per (b,c) channel: 64*64 = 4096 blocks, 4 waves each.
    welch_psd_wmma<<<dim3(64 * 64), dim3(128), SMEM_BYTES, stream>>>(x, out);
    (void)in_sizes; (void)n_in; (void)out_size; (void)d_ws; (void)ws_size;
}
